// LSTM_4071628997008
// MI455X (gfx1250) — matrix operand order repaired and checked by reading, not compiled, not run on hardware
//
#include <hip/hip_runtime.h>
#include <hip/hip_bf16.h>

// Problem constants (match reference)
#define VOCAB 32000
#define EMB   128
#define HID   256
#define NG    1024          // 4*HID gate columns
#define KTOT  384           // EMB + HID  (concatenated GEMM K)
#define TSEQ  512
#define BATCH 512
#define NCLS  20

#define MT      16          // batch rows per workgroup (one WMMA M tile)
#define WAVES   16          // wave w owns gate cols [16w,16w+16) of i,f,g,o
#define THREADS (WAVES * 32)

// LDS row stride for the A staging buffer, padded so that consecutive rows
// shift by 4 banks: 392 halfs = 196 dwords; 196 % 64 = 4. A 16-row b128
// (4-dword) access then covers all 64 LDS banks exactly once.
#define XH_STRIDE 392

typedef _Float16 f16;
typedef __attribute__((ext_vector_type(16))) _Float16 v16h;
typedef __attribute__((ext_vector_type(8)))  _Float16 v8h;
typedef __attribute__((ext_vector_type(8)))  float    v8f;

// Branch-free activations: v_exp_f32 (TRANS, co-executes with WMMA of other
// waves) + v_rcp_f32. No libm exec-mask divergence on the critical path.
__device__ __forceinline__ float fast_sigmoid(float x) {
    // exp(-x) -> inf for very negative x: rcp(inf) = 0, correct limit.
    return __builtin_amdgcn_rcpf(1.f + __expf(-x));
}
__device__ __forceinline__ float fast_tanh(float x) {
    // clamp so exp(2x) stays finite (tanh is +/-1 to f32 precision by |x|=10)
    const float xc = fminf(fmaxf(x, -15.f), 15.f);
    const float t  = __expf(2.f * xc);
    return (t - 1.f) * __builtin_amdgcn_rcpf(t + 1.f);
}

// ---------------------------------------------------------------------------
// Prep: Wcat[g][k] = f16( k<EMB ? W_ih[g][k] : W_hh[g][k-EMB] ),
//       bias[g]   = b_ih[g] + b_hh[g]
// ---------------------------------------------------------------------------
__global__ __launch_bounds__(KTOT) void prep_weights(
    const float* __restrict__ W_ih, const float* __restrict__ W_hh,
    const float* __restrict__ b_ih, const float* __restrict__ b_hh,
    f16* __restrict__ wcat, float* __restrict__ bias)
{
    const int g = blockIdx.x;      // 0..1023
    const int j = threadIdx.x;     // 0..383
    const float v = (j < EMB) ? W_ih[g * EMB + j] : W_hh[g * HID + (j - EMB)];
    wcat[g * KTOT + j] = (f16)v;
    if (j == 0) bias[g] = b_ih[g] + b_hh[g];
}

// ---------------------------------------------------------------------------
// Persistent LSTM kernel: 16 batch rows / WG, whole T loop inside.
// A operand = [x_t | h] (16 x 384 f16) staged in LDS; c,h live in VGPRs.
// Wave w owns gate columns [16w,16w+16) of each of i/f/g/o, so the WMMA
// C-layout puts matching i/f/g/o elements in the same lane and the whole
// LSTM cell update is in-register. Two barriers per step.
// ---------------------------------------------------------------------------
__global__ __launch_bounds__(THREADS) void lstm_kernel(
    const int*  __restrict__ inputs,   // (B,T)
    const int*  __restrict__ lengths,  // (B,)
    const float* __restrict__ emb,     // (VOCAB,EMB)
    const f16*  __restrict__ wcat,     // (NG,KTOT) f16
    const float* __restrict__ bias,    // (NG,)
    const float* __restrict__ W_lin,   // (NCLS,HID)
    const float* __restrict__ b_lin,   // (NCLS,)
    float* __restrict__ out)           // (B,NCLS) log-softmax
{
    __shared__ f16   xh[MT][XH_STRIDE]; // WMMA A staging: x_t | h (padded)
    __shared__ float hf[MT][HID];       // final hidden, f32
    __shared__ float lbuf[MT][NCLS];    // logits

    const int tid    = threadIdx.x;
    const int lane   = tid & 31;
    const int w      = tid >> 5;       // wave id 0..15 (== gather row)
    const int lane16 = lane & 15;
    const int laneHi = lane >> 4;      // 0|1 (WMMA half-wave)
    const int b0     = blockIdx.x * MT;

    // zero the h region of the A staging buffer
    for (int i = tid; i < MT * HID; i += THREADS) {
        xh[i / HID][EMB + (i % HID)] = (f16)0.f;
    }

    // per-lane bias broadcast: gate g, column 16w + lane16
    float biasr[4];
#pragma unroll
    for (int g = 0; g < 4; ++g) biasr[g] = bias[g * HID + 16 * w + lane16];

    // sequence lengths for the 8 rows this lane's C elements belong to
    int len_r[8];
#pragma unroll
    for (int e = 0; e < 8; ++e) len_r[e] = lengths[b0 + e + 8 * laneHi];

    // c/h state in registers: element e + 8*laneHi = row, column = 16w+lane16
    v8f c_reg, h_reg;
#pragma unroll
    for (int e = 0; e < 8; ++e) { c_reg[e] = 0.f; h_reg[e] = 0.f; }

    for (int t = 0; t < TSEQ; ++t) {
        // embedding gather -> xh[w][0:128] as f16. All lanes of wave w load
        // the same token (coalesced broadcast), then one float4 per lane
        // covers the whole 512-byte embedding row.
        {
            const int tokr = inputs[(b0 + w) * TSEQ + t];
            const int c = lane * 4;
            const float4 v = *(const float4*)(emb + (long long)tokr * EMB + c);
            xh[w][c + 0] = (f16)v.x; xh[w][c + 1] = (f16)v.y;
            xh[w][c + 2] = (f16)v.z; xh[w][c + 3] = (f16)v.w;
        }
        __syncthreads();   // x_t (this step) and h (prev step) visible to all

        // gates = [x|h] @ Wcat^T + bias   (16x384 @ 384x(4x16) per wave)
        v8f acc[4];
#pragma unroll
        for (int g = 0; g < 4; ++g)
#pragma unroll
            for (int e = 0; e < 8; ++e) acc[g][e] = biasr[g];

        // Unroll capped at 2: keeps the B-fragment addresses loop-varying so
        // LICM cannot hoist the (t-invariant) weight loads out of the T loop
        // and spill them to scratch. Loads stay as in-loop global_load_b128.
#pragma unroll 2
        for (int k = 0; k < KTOT; k += 32) {
            // ---- A fragment (16x32 f16): lanes 0-15 row M=lane, K {0..7,16..23};
            //      lanes 16-31 row M=lane-16, K {8..15,24..31}
            const int aoff = k + (laneHi ? 8 : 0);
            const v8h alo = *(const v8h*)&xh[lane16][aoff];
            const v8h ahi = *(const v8h*)&xh[lane16][aoff + 16];
            v16h a;
#pragma unroll
            for (int i = 0; i < 8; ++i) { a[i] = alo[i]; a[i + 8] = ahi[i]; }

#pragma unroll
            for (int g = 0; g < 4; ++g) {
                // ---- B fragment (32x16 f16): lane holds column n = lane16,
                //      K {0..7,16..23} (lanes 0-15) / {8..15,24..31} (lanes 16-31), matching A.
                //      Non-temporal: this 768 KB/step stream would only thrash
                //      the 64 KB WGP$; it lives in L2 (192 MB) anyway.
                const f16* brow = wcat
                    + (long long)(g * HID + 16 * w + lane16) * KTOT
                    + k + (laneHi ? 8 : 0);
                const v8h bl = __builtin_nontemporal_load((const v8h*)brow);
                const v8h bh = __builtin_nontemporal_load((const v8h*)(brow + 16));
                v16h bb;
#pragma unroll
                for (int i = 0; i < 8; ++i) { bb[i] = bl[i]; bb[i + 8] = bh[i]; }

                acc[g] = __builtin_amdgcn_wmma_f32_16x16x32_f16(
                    false, a, false, bb, (short)0, acc[g], false, false);
            }
        }
        __syncthreads();   // all xh reads done before h/x are overwritten

        // LSTM cell, fully in-register; masked by sequence length
#pragma unroll
        for (int e = 0; e < 8; ++e) {
            const int r = e + 8 * laneHi;      // row in tile
            const float iv = fast_sigmoid(acc[0][e]);
            const float fv = fast_sigmoid(acc[1][e]);
            const float gv = fast_tanh(acc[2][e]);
            const float ov = fast_sigmoid(acc[3][e]);
            const float cn = fv * c_reg[e] + iv * gv;
            const float hn = ov * fast_tanh(cn);
            const bool  m  = (t < len_r[e]);
            c_reg[e] = m ? cn : c_reg[e];
            h_reg[e] = m ? hn : h_reg[e];
            xh[r][EMB + 16 * w + lane16] = (f16)h_reg[e];   // next-step A operand
        }
    }

    // publish final hidden state (f32) for the classifier
#pragma unroll
    for (int e = 0; e < 8; ++e) {
        hf[e + 8 * laneHi][16 * w + lane16] = h_reg[e];
    }
    __syncthreads();

    // logits = h @ W_lin^T + b_lin   (16x256 @ 256x20, tiny)
    for (int idx = tid; idx < MT * NCLS; idx += THREADS) {
        const int r = idx / NCLS, c = idx % NCLS;
        float a = b_lin[c];
        for (int j = 0; j < HID; ++j) a += hf[r][j] * W_lin[c * HID + j];
        lbuf[r][c] = a;
    }
    __syncthreads();

    // log_softmax per row
    if (tid < MT) {
        const int r = tid;
        float mx = lbuf[r][0];
        for (int c = 1; c < NCLS; ++c) mx = fmaxf(mx, lbuf[r][c]);
        float s = 0.f;
        for (int c = 0; c < NCLS; ++c) s += __expf(lbuf[r][c] - mx);
        const float lse = mx + __logf(s);
        for (int c = 0; c < NCLS; ++c) out[(b0 + r) * NCLS + c] = lbuf[r][c] - lse;
    }
}

// ---------------------------------------------------------------------------
extern "C" void kernel_launch(void* const* d_in, const int* in_sizes, int n_in,
                              void* d_out, int out_size, void* d_ws, size_t ws_size,
                              hipStream_t stream)
{
    const int*   inputs  = (const int*)d_in[0];
    const int*   lengths = (const int*)d_in[1];
    const float* emb     = (const float*)d_in[2];
    const float* W_ih    = (const float*)d_in[3];
    const float* W_hh    = (const float*)d_in[4];
    const float* b_ih    = (const float*)d_in[5];
    const float* b_hh    = (const float*)d_in[6];
    const float* W_lin   = (const float*)d_in[7];
    const float* b_lin   = (const float*)d_in[8];
    float*       out     = (float*)d_out;

    // workspace: f16 Wcat (1024x384 = 768 KB) + combined bias (4 KB)
    f16*   wcat = (f16*)d_ws;
    float* bias = (float*)((char*)d_ws + (size_t)NG * KTOT * sizeof(f16));

    prep_weights<<<NG, KTOT, 0, stream>>>(W_ih, W_hh, b_ih, b_hh, wcat, bias);
    lstm_kernel<<<BATCH / MT, THREADS, 0, stream>>>(
        inputs, lengths, emb, wcat, bias, W_lin, b_lin, out);
}
